// TokenizerEncoder_44427141710574
// MI455X (gfx1250) — matrix operand order repaired and checked by reading, not compiled, not run on hardware
//
#include <hip/hip_runtime.h>

#define Bn 4
#define Sn 2048
#define Dn 512
#define Fn 2048
#define Hn 8
#define DHn 64
#define Mn (Bn*Sn)   // 8192 rows
#define Ln 2

// ---- output layout (floats) ----
#define OFF_XT   0
#define OFF_IDS  4194304              // Mn*Dn
#define OFF_AVG  4202496              // + Mn
#define OFF_MASK 4202497              // + 1
#define OFF_CLL  20979713             // + Bn*Sn*Sn

typedef _Float16 v16h __attribute__((ext_vector_type(16)));
typedef _Float16 v8h  __attribute__((ext_vector_type(8)));
typedef float    v8f  __attribute__((ext_vector_type(8)));

__device__ __forceinline__ v8f wmma16(v16h a, v16h b, v8f c) {
  return __builtin_amdgcn_wmma_f32_16x16x32_f16(false, a, false, b, (short)0, c,
                                                false, false);
}

// A fragment: 16x32 f16, M = lane&15, K = k0 + (e/8)*16 + (lane>>4)*8 + e%8
__device__ __forceinline__ v16h load_a_frag(const _Float16* base, int ld, int row,
                                            int k0, int lane) {
  const _Float16* p = base + (size_t)row * ld + k0 + ((lane >> 4) << 3);
  union { v16h v; v8h h[2]; } u;
  u.h[0] = *(const v8h*)(p);
  u.h[1] = *(const v8h*)(p + 16);
  return u.v;
}

// B fragment: 32x16 f16, N = lane&15, K = k0 + (e/8)*16 + (lane>>4)*8 + e%8
__device__ __forceinline__ v16h load_b_frag(const _Float16* base, int ld, int col,
                                            int k0, int lane) {
  const _Float16* p = base + (size_t)col * ld + k0 + ((lane >> 4) << 3);
  union { v16h v; v8h h[2]; } u;
  u.h[0] = *(const v8h*)(p);
  u.h[1] = *(const v8h*)(p + 16);
  return u.v;
}

// ---------------- weight cast+transpose: W (KxN f32) -> Wt (NxK f16) -------------
__global__ void __launch_bounds__(256) castWT_kernel(const float* __restrict__ W,
                                                     _Float16* __restrict__ Wt,
                                                     int K, int N) {
  int i = blockIdx.x * 256 + threadIdx.x;
  if (i >= K * N) return;
  int k = i / N, n = i - k * N;
  Wt[(size_t)n * K + k] = (_Float16)W[i];
}

// ---------------- RMSNorm: x (Mn x Dn f32) -> h (f16) ----------------------------
__global__ void __launch_bounds__(256) rms_kernel(const float* __restrict__ x,
                                                  const float* __restrict__ w,
                                                  _Float16* __restrict__ out) {
  int row = blockIdx.x;
  const float* xr = x + (size_t)row * Dn;
  float ss = 0.f;
  for (int i = threadIdx.x; i < Dn; i += 256) { float v = xr[i]; ss += v * v; }
  for (int d = 16; d > 0; d >>= 1) ss += __shfl_xor(ss, d);
  __shared__ float red[8];
  if ((threadIdx.x & 31) == 0) red[threadIdx.x >> 5] = ss;
  __syncthreads();
  if (threadIdx.x == 0) {
    float t = 0.f;
    for (int i = 0; i < 8; ++i) t += red[i];
    red[0] = t;
  }
  __syncthreads();
  float scale = rsqrtf(red[0] / (float)Dn + 1e-6f);
  for (int i = threadIdx.x; i < Dn; i += 256)
    out[(size_t)row * Dn + i] = (_Float16)(xr[i] * scale * w[i]);
}

// ---------------- generic WMMA GEMM: C = act(A*Wt) (+res), one wave / 16x16 tile -
__global__ void __launch_bounds__(256) gemm_kernel(const _Float16* __restrict__ A,
                                                   const _Float16* __restrict__ Bt,
                                                   const float* __restrict__ res,
                                                   float* __restrict__ outf,
                                                   _Float16* __restrict__ outh,
                                                   int M, int K, int N, int act) {
  int gid = blockIdx.x * 256 + threadIdx.x;
  int wave = gid >> 5, lane = gid & 31;
  int ntiles = N >> 4;
  int tm = wave / ntiles, tn = wave - tm * ntiles;
  if (tm * 16 >= M) return;
  int l16 = lane & 15;
  int arow = tm * 16 + l16;
  int bcol = tn * 16 + l16;
  v8f c = {};
  for (int k0 = 0; k0 < K; k0 += 32) {
    v16h a = load_a_frag(A, K, arow, k0, lane);
    v16h b = load_b_frag(Bt, K, bcol, k0, lane);
    c = wmma16(a, b, c);
  }
  int mbase = tm * 16 + ((lane >> 4) << 3);
  int n = tn * 16 + l16;
#pragma unroll
  for (int r = 0; r < 8; ++r) {
    float v = c[r];
    if (act) v = 0.5f * v * (1.0f + erff(v * 0.70710678118f));  // exact gelu
    size_t idx = (size_t)(mbase + r) * N + n;
    if (res) v += res[idx];
    if (outf) outf[idx] = v;
    if (outh) outh[idx] = (_Float16)v;
  }
}

// ---------------- RoPE + pack q,k -> (B,H,S,Dh) f16 ; v -> (B,H,Dh,S) f16 --------
__global__ void __launch_bounds__(256) rope_kernel(const float* __restrict__ q,
                                                   const float* __restrict__ k,
                                                   const float* __restrict__ v,
                                                   _Float16* __restrict__ qo,
                                                   _Float16* __restrict__ ko,
                                                   _Float16* __restrict__ vt) {
  int idx = blockIdx.x * 256 + threadIdx.x;       // Bn*Hn*Sn*32 threads
  int i = idx & 31;
  int s = (idx >> 5) & (Sn - 1);
  int h = (idx >> 16) & (Hn - 1);
  int b = idx >> 19;
  size_t rowoff = ((size_t)(b * Sn + s)) * Dn + h * DHn;
  size_t off2 = (((size_t)(b * Hn + h)) * Sn + s) * DHn;
  float fr = (float)s * powf(10000.0f, -(float)i / 32.0f);
  float cs = cosf(fr), sn = sinf(fr);
  float q1 = q[rowoff + i], q2 = q[rowoff + 32 + i];
  qo[off2 + i]      = (_Float16)(q1 * cs - q2 * sn);
  qo[off2 + 32 + i] = (_Float16)(q1 * sn + q2 * cs);
  float k1 = k[rowoff + i], k2 = k[rowoff + 32 + i];
  ko[off2 + i]      = (_Float16)(k1 * cs - k2 * sn);
  ko[off2 + 32 + i] = (_Float16)(k1 * sn + k2 * cs);
  size_t vbase = ((size_t)(b * Hn + h)) * DHn;
  vt[(vbase + i) * Sn + s]      = (_Float16)v[rowoff + i];
  vt[(vbase + 32 + i) * Sn + s] = (_Float16)v[rowoff + 32 + i];
}

// ---------------- flash attention: one wave per (b,h,16-query tile) --------------
__global__ void __launch_bounds__(32) attn_kernel(const _Float16* __restrict__ Q,
                                                  const _Float16* __restrict__ Km,
                                                  const _Float16* __restrict__ Vt,
                                                  _Float16* __restrict__ Out) {
  int lane = threadIdx.x;
  int tile = blockIdx.x;
  int qt = tile & (Sn / 16 - 1);
  int bh = tile >> 7;
  int b = bh >> 3, h = bh & 7;
  const _Float16* qp = Q + (size_t)bh * Sn * DHn;
  const _Float16* kp = Km + (size_t)bh * Sn * DHn;
  const _Float16* vp = Vt + (size_t)bh * DHn * Sn;
  int m0 = qt << 4;
  int l16 = lane & 15, lh = lane >> 4;

  v16h aq0 = load_a_frag(qp, DHn, m0 + l16, 0, lane);
  v16h aq1 = load_a_frag(qp, DHn, m0 + l16, 32, lane);

  float mrow[8], lsum[8];
  v8f o0 = {}, o1 = {}, o2 = {}, o3 = {};
#pragma unroll
  for (int r = 0; r < 8; ++r) { mrow[r] = -3.0e38f; lsum[r] = 0.f; }

  __shared__ _Float16 pls[16 * 32] __attribute__((aligned(64)));
  const float scale = 0.125f;  // 1/sqrt(64)
  int nkt = ((m0 + 15) >> 5) + 1;

  for (int kt = 0; kt < nkt; ++kt) {
    int n0 = kt << 5;
    v8f c0 = {}, c1 = {};
    {
      v16h bk = load_b_frag(kp, DHn, n0 + l16, 0, lane);       c0 = wmma16(aq0, bk, c0);
      bk = load_b_frag(kp, DHn, n0 + l16, 32, lane);           c0 = wmma16(aq1, bk, c0);
      bk = load_b_frag(kp, DHn, n0 + 16 + l16, 0, lane);       c1 = wmma16(aq0, bk, c1);
      bk = load_b_frag(kp, DHn, n0 + 16 + l16, 32, lane);      c1 = wmma16(aq1, bk, c1);
    }
    int nk0 = n0 + l16;
#pragma unroll
    for (int r = 0; r < 8; ++r) {
      int mq = m0 + (lh << 3) + r;
      float s0 = (nk0 <= mq)      ? c0[r] * scale : -3.0e38f;
      float s1 = (nk0 + 16 <= mq) ? c1[r] * scale : -3.0e38f;
      float mx = fmaxf(s0, s1);
      for (int d = 1; d < 16; d <<= 1) mx = fmaxf(mx, __shfl_xor(mx, d));
      float nm = fmaxf(mrow[r], mx);
      float alpha = __expf(mrow[r] - nm);
      mrow[r] = nm;
      float p0 = __expf(s0 - nm), p1 = __expf(s1 - nm);
      float rs = p0 + p1;
      for (int d = 1; d < 16; d <<= 1) rs += __shfl_xor(rs, d);
      lsum[r] = lsum[r] * alpha + rs;
      o0[r] *= alpha; o1[r] *= alpha; o2[r] *= alpha; o3[r] *= alpha;
      c0[r] = p0; c1[r] = p1;
    }
    // re-layout P (C-layout) -> A-layout through LDS
    __syncthreads();
#pragma unroll
    for (int r = 0; r < 8; ++r) {
      int mloc = (lh << 3) + r;
      pls[mloc * 32 + l16]      = (_Float16)c0[r];
      pls[mloc * 32 + 16 + l16] = (_Float16)c1[r];
    }
    __syncthreads();
    v16h ap = load_a_frag((const _Float16*)pls, 32, l16, 0, lane);
    {
      v16h bv = load_b_frag(vp, Sn, 0 + l16, n0, lane);   o0 = wmma16(ap, bv, o0);
      bv = load_b_frag(vp, Sn, 16 + l16, n0, lane);       o1 = wmma16(ap, bv, o1);
      bv = load_b_frag(vp, Sn, 32 + l16, n0, lane);       o2 = wmma16(ap, bv, o2);
      bv = load_b_frag(vp, Sn, 48 + l16, n0, lane);       o3 = wmma16(ap, bv, o3);
    }
    __syncthreads();
  }
#pragma unroll
  for (int r = 0; r < 8; ++r) {
    float inv = 1.0f / lsum[r];
    int mq = m0 + (lh << 3) + r;
    _Float16* op = Out + ((size_t)(b * Sn + mq)) * Dn + h * DHn + l16;
    op[0]  = (_Float16)(o0[r] * inv);
    op[16] = (_Float16)(o1[r] * inv);
    op[32] = (_Float16)(o2[r] * inv);
    op[48] = (_Float16)(o3[r] * inv);
  }
}

// ---------------- head: probs = sigmoid(x @ head_w + b), one wave per row --------
__global__ void __launch_bounds__(32) head_kernel(const float* __restrict__ x,
                                                  const float* __restrict__ hw,
                                                  const float* __restrict__ hb,
                                                  float* __restrict__ probs) {
  int row = blockIdx.x;
  int lane = threadIdx.x;
  const float* xr = x + (size_t)row * Dn;
  float s = 0.f;
  for (int i = lane; i < Dn; i += 32) s += xr[i] * hw[i];
  for (int d = 16; d > 0; d >>= 1) s += __shfl_xor(s, d);
  if (lane == 0) probs[row] = 1.0f / (1.0f + __expf(-(s + hb[0])));
}

// ---------------- chunk_len_loss = sum(probs)/B ----------------------------------
__global__ void __launch_bounds__(256) loss_kernel(const float* __restrict__ probs,
                                                   float* __restrict__ out) {
  __shared__ float red[256];
  float s = 0.f;
  for (int i = threadIdx.x; i < Mn; i += 256) s += probs[i];
  red[threadIdx.x] = s;
  __syncthreads();
  for (int o = 128; o > 0; o >>= 1) {
    if (threadIdx.x < o) red[threadIdx.x] += red[threadIdx.x + o];
    __syncthreads();
  }
  if (threadIdx.x == 0) *out = red[0] * (1.0f / (float)Bn);
}

// ---------------- sequential chunk ids, one thread per batch ---------------------
__global__ void chunk_kernel(const float* __restrict__ probs, int* __restrict__ ids,
                             int* __restrict__ nch) {
  int b = blockIdx.x;
  const float* p = probs + (size_t)b * Sn;
  __shared__ int nxt[Sn + 1];
  bool any = false;
  for (int s = 0; s < Sn; ++s) any = any || (p[s] > 0.5f);
  int mn = Sn + 1;
  for (int i = Sn; i >= 0; --i) {
    bool he = any ? (i >= 1 && p[i - 1] > 0.5f) : (i == Sn - 1);
    if (he && i < mn) mn = i;
    nxt[i] = mn;
  }
  int cur_end = 0, cid = -1;
  for (int t = 0; t < Sn; ++t) {
    if (t == cur_end) {
      int cap = t + 16; if (cap > Sn) cap = Sn;       // MAX_CL
      int e = nxt[t + 1];
      if (e > cap) e = cap;
      if (e - t < 3) { e = t + 3; if (e > Sn) e = Sn; } // MIN_CL
      cur_end = e; cid++;
    }
    ids[(size_t)b * Sn + t] = cid;
  }
  nch[b] = cid + 1;
}

__global__ void final_kernel(const int* __restrict__ nch, float* __restrict__ out) {
  float s = 0.f;
  for (int b = 0; b < Bn; ++b) s += (float)Sn / (float)nch[b];
  *out = s * (1.0f / (float)Bn);
}

__global__ void __launch_bounds__(256) mask_kernel(const int* __restrict__ ids,
                                                   float* __restrict__ out) {
  size_t i = (size_t)blockIdx.x * 256 + threadIdx.x;
  int b = (int)(i >> 22);
  int r = (int)((i >> 11) & (Sn - 1));
  int c = (int)(i & (Sn - 1));
  out[i] = (ids[(size_t)b * Sn + r] == ids[(size_t)b * Sn + c]) ? 1.0f : 0.0f;
}

__global__ void __launch_bounds__(256) copy_kernel(const float* __restrict__ src,
                                                   float* __restrict__ dst, int n) {
  int i = blockIdx.x * 256 + threadIdx.x;
  if (i < n) dst[i] = src[i];
}

__global__ void __launch_bounds__(256) castids_kernel(const int* __restrict__ src,
                                                      float* __restrict__ dst, int n) {
  int i = blockIdx.x * 256 + threadIdx.x;
  if (i < n) dst[i] = (float)src[i];
}

// =================================================================================
extern "C" void kernel_launch(void* const* d_in, const int* in_sizes, int n_in,
                              void* d_out, int out_size, void* d_ws, size_t ws_size,
                              hipStream_t stream) {
  (void)in_sizes; (void)n_in; (void)out_size; (void)ws_size;
  const float* x_in  = (const float*)d_in[0];
  const int*   xids  = (const int*)d_in[1];
  const float* ln1   = (const float*)d_in[2];
  const float* Wq    = (const float*)d_in[3];
  const float* Wk    = (const float*)d_in[4];
  const float* Wv    = (const float*)d_in[5];
  const float* Wo    = (const float*)d_in[6];
  const float* ln2   = (const float*)d_in[7];
  const float* W1    = (const float*)d_in[8];
  const float* W2    = (const float*)d_in[9];
  const float* headw = (const float*)d_in[10];
  const float* headb = (const float*)d_in[11];
  float* out = (float*)d_out;

  char* w = (char*)d_ws;
  size_t off = 0;
  auto alloc = [&](size_t bytes) -> void* {
    void* p = (void*)(w + off);
    off += (bytes + 255) & ~(size_t)255;
    return p;
  };
  float*    xb   = (float*)alloc((size_t)Mn * Dn * 4);
  _Float16* h16  = (_Float16*)alloc((size_t)Mn * Dn * 2);
  float*    qf   = (float*)alloc((size_t)Mn * Dn * 4);
  float*    kf   = (float*)alloc((size_t)Mn * Dn * 4);
  float*    vf   = (float*)alloc((size_t)Mn * Dn * 4);
  _Float16* g16  = (_Float16*)qf;  // alias: q/k f32 dead by FFN time (32MB)
  _Float16* q16  = (_Float16*)alloc((size_t)Mn * Dn * 2);
  _Float16* k16  = (_Float16*)alloc((size_t)Mn * Dn * 2);
  _Float16* vt16 = (_Float16*)alloc((size_t)Mn * Dn * 2);
  _Float16* a16  = (_Float16*)alloc((size_t)Mn * Dn * 2);
  _Float16* wqt  = (_Float16*)alloc((size_t)Ln * Dn * Dn * 2);
  _Float16* wkt  = (_Float16*)alloc((size_t)Ln * Dn * Dn * 2);
  _Float16* wvt  = (_Float16*)alloc((size_t)Ln * Dn * Dn * 2);
  _Float16* wot  = (_Float16*)alloc((size_t)Ln * Dn * Dn * 2);
  _Float16* w1t  = (_Float16*)alloc((size_t)Ln * Dn * Fn * 2);
  _Float16* w2t  = (_Float16*)alloc((size_t)Ln * Fn * Dn * 2);
  float*    probs = (float*)alloc((size_t)Mn * 4);
  int*      ids   = (int*)alloc((size_t)Mn * 4);
  int*      nch   = (int*)alloc(256);

  // ---- convert & transpose weights to f16 ----
  const int nDD = Dn * Dn, nDF = Dn * Fn;
  for (int l = 0; l < Ln; ++l) {
    castWT_kernel<<<(nDD + 255) / 256, 256, 0, stream>>>(Wq + (size_t)l * nDD, wqt + (size_t)l * nDD, Dn, Dn);
    castWT_kernel<<<(nDD + 255) / 256, 256, 0, stream>>>(Wk + (size_t)l * nDD, wkt + (size_t)l * nDD, Dn, Dn);
    castWT_kernel<<<(nDD + 255) / 256, 256, 0, stream>>>(Wv + (size_t)l * nDD, wvt + (size_t)l * nDD, Dn, Dn);
    castWT_kernel<<<(nDD + 255) / 256, 256, 0, stream>>>(Wo + (size_t)l * nDD, wot + (size_t)l * nDD, Dn, Dn);
    castWT_kernel<<<(nDF + 255) / 256, 256, 0, stream>>>(W1 + (size_t)l * nDF, w1t + (size_t)l * nDF, Dn, Fn);
    castWT_kernel<<<(nDF + 255) / 256, 256, 0, stream>>>(W2 + (size_t)l * nDF, w2t + (size_t)l * nDF, Fn, Dn);
  }

  hipMemcpyAsync(xb, x_in, (size_t)Mn * Dn * 4, hipMemcpyDeviceToDevice, stream);

  const int blkDD = (Mn / 16) * (Dn / 16) * 32 / 256;  // 2048 blocks
  const int blkDF = (Mn / 16) * (Fn / 16) * 32 / 256;  // 8192 blocks
  for (int l = 0; l < Ln; ++l) {
    rms_kernel<<<Mn, 256, 0, stream>>>(xb, ln1 + l * Dn, h16);
    gemm_kernel<<<blkDD, 256, 0, stream>>>(h16, wqt + (size_t)l * nDD, nullptr, qf, nullptr, Mn, Dn, Dn, 0);
    gemm_kernel<<<blkDD, 256, 0, stream>>>(h16, wkt + (size_t)l * nDD, nullptr, kf, nullptr, Mn, Dn, Dn, 0);
    gemm_kernel<<<blkDD, 256, 0, stream>>>(h16, wvt + (size_t)l * nDD, nullptr, vf, nullptr, Mn, Dn, Dn, 0);
    rope_kernel<<<(Bn * Hn * Sn * 32) / 256, 256, 0, stream>>>(qf, kf, vf, q16, k16, vt16);
    attn_kernel<<<Bn * Hn * (Sn / 16), 32, 0, stream>>>(q16, k16, vt16, a16);
    gemm_kernel<<<blkDD, 256, 0, stream>>>(a16, wot + (size_t)l * nDD, xb, xb, nullptr, Mn, Dn, Dn, 0);
    rms_kernel<<<Mn, 256, 0, stream>>>(xb, ln2 + l * Dn, h16);
    gemm_kernel<<<blkDF, 256, 0, stream>>>(h16, w1t + (size_t)l * nDF, nullptr, nullptr, g16, Mn, Dn, Fn, 1);
    gemm_kernel<<<blkDD, 256, 0, stream>>>(g16, w2t + (size_t)l * nDF, xb, xb, nullptr, Mn, Fn, Dn, 0);
  }

  head_kernel<<<Mn, 32, 0, stream>>>(xb, headw, headb, probs);
  loss_kernel<<<1, 256, 0, stream>>>(probs, out + OFF_CLL);
  chunk_kernel<<<Bn, 1, 0, stream>>>(probs, ids, nch);
  final_kernel<<<1, 1, 0, stream>>>(nch, out + OFF_AVG);
  mask_kernel<<<(Bn * Sn * Sn) / 256, 256, 0, stream>>>(ids, out + OFF_MASK);
  copy_kernel<<<(Mn * Dn) / 256, 256, 0, stream>>>(xb, out + OFF_XT, Mn * Dn);
  castids_kernel<<<(Mn + 255) / 256, 256, 0, stream>>>(xids, out + OFF_IDS, Mn);
}
